// LMLTVIT_62646392979895
// MI455X (gfx1250) — compile-verified, matrix operand order checked by reading, not run on hardware
//
#include <hip/hip_runtime.h>

typedef __attribute__((ext_vector_type(16))) __bf16 bf16x16;
typedef __attribute__((ext_vector_type(8)))  __bf16 bf16x8;
typedef __attribute__((ext_vector_type(8)))  float  fx8;

#define C_DIM 96
#define TOK   64
#define IMG   256
#define SCALE 0.102062072615966f   // 96^-0.5

// ---------- WMMA helpers (CDNA5 gfx1250, wave32) ----------

__device__ __forceinline__ fx8 wmma_bf16(bf16x16 a, bf16x16 b, fx8 c) {
  return __builtin_amdgcn_wmma_f32_16x16x32_bf16(false, a, false, b, (short)0, c,
                                                 false, false);
}

// Load one 16x32 bf16 operand (A-layout; used for both A and N-major B) from LDS.
// Layout (ISA 7.12.2): lanes 0-15 row=lane, halves 0..7 -> K 0..7, 8..15 -> K 16..23;
// lanes 16-31 row=lane-16, halves 0..7 -> K 8..15, 8..15 -> K 24..31.
__device__ __forceinline__ bf16x16 ld_ab(const __bf16* base, int stride, int lane) {
  const int r = lane & 15;
  const int khi = (lane >> 4) & 1;
  const __bf16* p = base + r * stride + 8 * khi;
  bf16x8 lo = *(const bf16x8*)p;
  bf16x8 hi = *(const bf16x8*)(p + 16);
  bf16x16 o;
#pragma unroll
  for (int i = 0; i < 8; ++i) { o[i] = lo[i]; o[i + 8] = hi[i]; }
  return o;
}

// Load a pre-packed B-operand tile: 32 lanes x 16 halves, lane-contiguous (32B/lane).
__device__ __forceinline__ bf16x16 ld_bp(const __bf16* packed, int tix, int lane) {
  return *(const bf16x16*)(packed + (tix * 32 + lane) * 16);
}

// ---------- Pack kernel: w_comb + bf16 WMMA-tiled weights ----------
// d_ws layout: [0)            wcomb  f32 96*25
//              [16384)        qkvp   bf16 18*3*32*16  (55296 B)
//              [73728)        projp  bf16  6*3*32*16  (18432 B)

__global__ void lmlt_pack(const float* __restrict__ alpha,
                          const float* __restrict__ w1,
                          const float* __restrict__ w3,
                          const float* __restrict__ w5,
                          const float* __restrict__ qkvw,
                          const float* __restrict__ projw,
                          float* __restrict__ wcomb,
                          __bf16* __restrict__ qkvp,
                          __bf16* __restrict__ projp) {
  int tid = blockIdx.x * 256 + threadIdx.x;

  if (tid < 96 * 25) {               // combined 5x5 depthwise kernel
    int c = tid / 25, ij = tid % 25, i = ij / 5, j = ij % 5;
    float v = alpha[3] * w5[c * 25 + ij];
    if (i >= 1 && i <= 3 && j >= 1 && j <= 3)
      v += alpha[2] * w3[c * 9 + (i - 1) * 3 + (j - 1)];
    if (i == 2 && j == 2) v += alpha[0] + alpha[1] * w1[c];
    wcomb[tid] = v;
  }

  int q = tid - 4096;                // qkv_w (288x96) -> 18 ntiles x 3 ksteps
  if (q >= 0 && q < 18 * 3 * 512) {
    int h = q & 15, lane = (q >> 4) & 31, rest = q >> 9;
    int ks = rest % 3, nt = rest / 3;
    int khi = (lane >> 4) & 1;
    int k = ks * 32 + ((h < 8) ? (8 * khi + h) : (16 + 8 * khi + (h - 8)));
    int n = nt * 16 + (lane & 15);
    qkvp[q] = (__bf16)qkvw[n * 96 + k];
  }

  int p = tid - 36864;               // proj_w (96x96) -> 6 ntiles x 3 ksteps
  if (p >= 0 && p < 6 * 3 * 512) {
    int h = p & 15, lane = (p >> 4) & 31, rest = p >> 9;
    int ks = rest % 3, nt = rest / 3;
    int khi = (lane >> 4) & 1;
    int k = ks * 32 + ((h < 8) ? (8 * khi + h) : (16 + 8 * khi + (h - 8)));
    int n = nt * 16 + (lane & 15);
    projp[p] = (__bf16)projw[n * 96 + k];
  }
}

// ---------- Fused per-window kernel ----------
// One block per 8x8 window; 256 threads = 8 wave32.
// LDS (96 KB dynamic):
//   sX  bf16 64x96  (conv out; reused as sP 64x64 for probs)
//   sQ  bf16 64x96  (scaled q; reused as sOb for proj A operand)
//   sK  bf16 64x96
//   sVt bf16 96x64  (v channel-major, B operand of P@V)
//   sV  f32  64x96  (v for LePE)
//   sF  f32  64x96  (scores 64x64, then attn-out 64x96)

__global__ void __launch_bounds__(256)
lmlt_attn(const float* __restrict__ x,
          const float* __restrict__ qkv_b,
          const float* __restrict__ proj_b,
          const float* __restrict__ getv_w,
          const float* __restrict__ getv_b,
          const float* __restrict__ wcomb,
          const __bf16* __restrict__ qkvp,
          const __bf16* __restrict__ projp,
          float* __restrict__ out) {
  extern __shared__ char smem[];
  __bf16* sX  = (__bf16*)smem;
  __bf16* sQ  = sX + TOK * C_DIM;
  __bf16* sK  = sQ + TOK * C_DIM;
  __bf16* sVt = sK + TOK * C_DIM;
  float*  sV  = (float*)(sVt + C_DIM * TOK);
  float*  sF  = sV + TOK * C_DIM;
  __bf16* sP  = sX;      // reuse after stage 2
  __bf16* sOb = sQ;      // reuse after stage 3

  const int tid  = threadIdx.x;
  const int wave = tid >> 5;
  const int lane = tid & 31;
  const int nl   = lane & 15;
  const int khi  = (lane >> 4) & 1;

  const int win = blockIdx.x;
  const int b   = win >> 10;
  const int h0  = ((win >> 5) & 31) * 8;
  const int w0  = (win & 31) * 8;

  // ---- Stage 1: 5x5 depthwise conv (with image halo) -> sX (bf16) ----
  for (int e = tid; e < TOK * C_DIM; e += 256) {
    int t = e / C_DIM, c = e - t * C_DIM;
    int hy = h0 + (t >> 3), wx = w0 + (t & 7);
    float acc = 0.f;
#pragma unroll
    for (int dy = 0; dy < 5; ++dy) {
      int hh = hy + dy - 2;
      if ((unsigned)hh >= (unsigned)IMG) continue;
#pragma unroll
      for (int dx = 0; dx < 5; ++dx) {
        int ww = wx + dx - 2;
        if ((unsigned)ww >= (unsigned)IMG) continue;
        acc += x[((size_t)b * (IMG * IMG) + hh * IMG + ww) * C_DIM + c] *
               wcomb[c * 25 + dy * 5 + dx];
      }
    }
    sX[t * C_DIM + c] = (__bf16)acc;
  }
  __syncthreads();

  // ---- Stage 2: QKV = sX @ qkv_w^T + b  (M=64, N=288, K=96) ----
  for (int nt = wave; nt < 18; nt += 8) {
    bf16x16 b0 = ld_bp(qkvp, nt * 3 + 0, lane);
    bf16x16 b1 = ld_bp(qkvp, nt * 3 + 1, lane);
    bf16x16 b2 = ld_bp(qkvp, nt * 3 + 2, lane);
    float bias = qkv_b[nt * 16 + nl];
#pragma unroll
    for (int mt = 0; mt < 4; ++mt) {
      const __bf16* arow = sX + mt * 16 * C_DIM;
      bf16x16 a0 = ld_ab(arow +  0, C_DIM, lane);
      bf16x16 a1 = ld_ab(arow + 32, C_DIM, lane);
      bf16x16 a2 = ld_ab(arow + 64, C_DIM, lane);
      fx8 acc = {};
      acc = wmma_bf16(a0, b0, acc);
      acc = wmma_bf16(a1, b1, acc);
      acc = wmma_bf16(a2, b2, acc);
      if (nt < 6) {                       // q (pre-scaled)
        int n = nt * 16 + nl;
#pragma unroll
        for (int i = 0; i < 8; ++i) {
          int m = mt * 16 + i + 8 * khi;
          sQ[m * C_DIM + n] = (__bf16)((acc[i] + bias) * SCALE);
        }
      } else if (nt < 12) {               // k
        int n = (nt - 6) * 16 + nl;
#pragma unroll
        for (int i = 0; i < 8; ++i) {
          int m = mt * 16 + i + 8 * khi;
          sK[m * C_DIM + n] = (__bf16)(acc[i] + bias);
        }
      } else {                            // v (f32 + transposed bf16)
        int c = (nt - 12) * 16 + nl;
#pragma unroll
        for (int i = 0; i < 8; ++i) {
          int m = mt * 16 + i + 8 * khi;
          float v = acc[i] + bias;
          sV[m * C_DIM + c] = v;
          sVt[c * TOK + m]  = (__bf16)v;
        }
      }
    }
  }
  __syncthreads();

  // ---- Stage 3: S = q @ k^T  (M=64, N=64, K=96) -> sF ----
  for (int t = wave; t < 16; t += 8) {
    int mt = t >> 2, ntt = t & 3;
    fx8 acc = {};
#pragma unroll
    for (int ks = 0; ks < 3; ++ks) {
      bf16x16 a  = ld_ab(sQ + mt  * 16 * C_DIM + ks * 32, C_DIM, lane);
      bf16x16 bb = ld_ab(sK + ntt * 16 * C_DIM + ks * 32, C_DIM, lane);
      acc = wmma_bf16(a, bb, acc);
    }
#pragma unroll
    for (int i = 0; i < 8; ++i)
      sF[(mt * 16 + i + 8 * khi) * TOK + ntt * 16 + nl] = acc[i];
  }
  __syncthreads();

  // ---- Stage 4: row softmax -> sP (bf16) ----
  if (tid < TOK) {
    float* row = sF + tid * TOK;
    float mx = -3.4e38f;
#pragma unroll 8
    for (int j = 0; j < TOK; ++j) mx = fmaxf(mx, row[j]);
    float sum = 0.f;
#pragma unroll 8
    for (int j = 0; j < TOK; ++j) { float e = __expf(row[j] - mx); row[j] = e; sum += e; }
    float inv = 1.f / sum;
    __bf16* pr = sP + tid * TOK;
#pragma unroll 8
    for (int j = 0; j < TOK; ++j) pr[j] = (__bf16)(row[j] * inv);
  }
  __syncthreads();

  // ---- Stage 5: O = P @ V  (M=64, N=96, K=64) -> sF ----
  for (int t = wave; t < 24; t += 8) {
    int mt = t / 6, nt2 = t % 6;
    fx8 acc = {};
#pragma unroll
    for (int ks = 0; ks < 2; ++ks) {
      bf16x16 a  = ld_ab(sP  + mt  * 16 * TOK + ks * 32, TOK, lane);
      bf16x16 bb = ld_ab(sVt + nt2 * 16 * TOK + ks * 32, TOK, lane);
      acc = wmma_bf16(a, bb, acc);
    }
#pragma unroll
    for (int i = 0; i < 8; ++i)
      sF[(mt * 16 + i + 8 * khi) * C_DIM + nt2 * 16 + nl] = acc[i];
  }
  __syncthreads();

  // ---- Stage 6: LePE 3x3 dwconv on v (window-local) + add -> sOb (bf16) ----
  for (int e = tid; e < TOK * C_DIM; e += 256) {
    int t = e / C_DIM, c = e - t * C_DIM;
    int y = t >> 3, xq = t & 7;
    float l = getv_b[c];
#pragma unroll
    for (int dy = 0; dy < 3; ++dy) {
      int yy = y + dy - 1;
      if ((unsigned)yy >= 8u) continue;
#pragma unroll
      for (int dx = 0; dx < 3; ++dx) {
        int xx = xq + dx - 1;
        if ((unsigned)xx >= 8u) continue;
        l += sV[(yy * 8 + xx) * C_DIM + c] * getv_w[c * 9 + dy * 3 + dx];
      }
    }
    sOb[e] = (__bf16)(sF[e] + l);
  }
  __syncthreads();

  // ---- Stage 7: proj + bias + window-reverse store (M=64, N=96, K=96) ----
  for (int t = wave; t < 24; t += 8) {
    int mt = t / 6, nt2 = t % 6;
    fx8 acc = {};
#pragma unroll
    for (int ks = 0; ks < 3; ++ks) {
      bf16x16 a  = ld_ab(sOb + mt * 16 * C_DIM + ks * 32, C_DIM, lane);
      bf16x16 bb = ld_bp(projp, nt2 * 3 + ks, lane);
      acc = wmma_bf16(a, bb, acc);
    }
    int n = nt2 * 16 + nl;
    float bias = proj_b[n];
#pragma unroll
    for (int i = 0; i < 8; ++i) {
      int tok = mt * 16 + i + 8 * khi;
      int hy = h0 + (tok >> 3), wx = w0 + (tok & 7);
      out[((size_t)b * (IMG * IMG) + hy * IMG + wx) * C_DIM + n] = acc[i] + bias;
    }
  }
}

// ---------- launch ----------

extern "C" void kernel_launch(void* const* d_in, const int* in_sizes, int n_in,
                              void* d_out, int out_size, void* d_ws, size_t ws_size,
                              hipStream_t stream) {
  const float* x     = (const float*)d_in[0];
  const float* alpha = (const float*)d_in[1];
  const float* w1    = (const float*)d_in[2];
  const float* w3    = (const float*)d_in[3];
  const float* w5    = (const float*)d_in[4];
  const float* qkvw  = (const float*)d_in[5];
  const float* qkvb  = (const float*)d_in[6];
  const float* projw = (const float*)d_in[7];
  const float* projb = (const float*)d_in[8];
  const float* getvw = (const float*)d_in[9];
  const float* getvb = (const float*)d_in[10];

  char* ws = (char*)d_ws;
  float*  wcomb = (float*)ws;
  __bf16* qkvp  = (__bf16*)(ws + 16384);
  __bf16* projp = (__bf16*)(ws + 73728);

  lmlt_pack<<<180, 256, 0, stream>>>(alpha, w1, w3, w5, qkvw, projw,
                                     wcomb, qkvp, projp);

  int Bn = in_sizes[0] / (IMG * IMG * C_DIM);   // = 8
  int nwin = Bn * 32 * 32;                      // 8x8 windows over 256x256
  size_t shmem = (size_t)(3 * TOK * C_DIM + C_DIM * TOK) * 2   // bf16 regions
               + (size_t)(2 * TOK * C_DIM) * 4;                // f32 regions
  lmlt_attn<<<nwin, 256, shmem, stream>>>(x, qkvb, projb, getvw, getvb,
                                          wcomb, qkvp, projp, (float*)d_out);
}
